// AttnBlock_62354335203559
// MI455X (gfx1250) — matrix operand order repaired and checked by reading, not compiled, not run on hardware
//
#include <hip/hip_runtime.h>
#include <hip/hip_bf16.h>

constexpr int BATCH = 4;
constexpr int CH    = 512;   // channels
constexpr int NPIX  = 4096;  // H*W
constexpr int NGRP  = 32;
constexpr int CPG   = 16;    // channels per group

typedef __bf16 v16bf __attribute__((ext_vector_type(16)));
typedef float  v8f   __attribute__((ext_vector_type(8)));
typedef unsigned int v4u __attribute__((ext_vector_type(4)));
typedef int v4i __attribute__((ext_vector_type(4)));
typedef int v8i __attribute__((ext_vector_type(8)));

struct __align__(16) U128 { unsigned int x[4]; };
union FragU { U128 u[2]; v16bf v; };

__device__ __forceinline__ unsigned short f2bf(float f) {
  unsigned int u = __float_as_uint(f);
  u += 0x7FFFu + ((u >> 16) & 1u);   // round-to-nearest-even
  return (unsigned short)(u >> 16);
}

__device__ __forceinline__ v8f wmma_bf16(v16bf a, v16bf b, v8f c) {
  return __builtin_amdgcn_wmma_f32_16x16x32_bf16(false, a, false, b, (short)0, c,
                                                 false, false);
}

// A fragment: 16x32 (MxK) row-major tile, leading dim ldk (elements).
// lane<16: K 0..15; lane>=16: K 16..31 (matches frag_b)
__device__ __forceinline__ v16bf frag_a(const unsigned short* base, int ldk, int lane) {
  int m = lane & 15, h = lane >> 4;
  FragU f;
  f.u[0] = *(const U128*)(base + (size_t)m * ldk + h * 16);
  f.u[1] = *(const U128*)(base + (size_t)m * ldk + h * 16 + 8);
  return f.v;
}
// B fragment: 32x16 (KxN) supplied as B^T row-major [N][K], leading dim ldk.
// lane<16: K 0..15; lane>=16: K 16..31 (one 32B contiguous run per lane)
__device__ __forceinline__ v16bf frag_b(const unsigned short* bt, size_t ldk, int lane) {
  int n = lane & 15, h = lane >> 4;
  const unsigned short* p = bt + (size_t)n * ldk + h * 16;
  FragU f;
  f.u[0] = *(const U128*)(p);
  f.u[1] = *(const U128*)(p + 8);
  return f.v;
}

// ---- Tensor Data Mover: 2-D bf16 tile (rows x cols) global -> LDS ----
// D# per CDNA5 ISA ch.8: group0 {count=1, lds_addr, global_addr[56:0], type=2},
// group1 {data_size=2B, tensor_dim0/1, tile_dim0/1, dim0_stride}, groups2/3 = 0.
#if defined(__clang_major__) && __clang_major__ >= 23
#define TDM_CALL(g0, g1, z0, z1) \
  __builtin_amdgcn_tensor_load_to_lds(g0, g1, z0, z1, (v8i){}, 0)
#else
#define TDM_CALL(g0, g1, z0, z1) \
  __builtin_amdgcn_tensor_load_to_lds(g0, g1, z0, z1, 0)
#endif

__device__ __forceinline__ void tdm_load_rows_bf16(const unsigned short* gsrc,
                                                   unsigned short* lds_dst,
                                                   unsigned rows, unsigned cols,
                                                   unsigned stride_elems) {
  unsigned long long ga = (unsigned long long)(uintptr_t)gsrc;
  v4u g0 = { 1u,                                        // count=1 (valid user D#)
             (unsigned)(uintptr_t)lds_dst,              // LDS byte address
             (unsigned)ga,                              // global addr [31:0]
             ((unsigned)(ga >> 32) & 0x01FFFFFFu) | 0x80000000u };  // [56:32] | type=2
  v8i g1 = { (int)0x00010000,                           // wg_mask=0, data_size=1 (2B)
             (int)((cols & 0xFFFFu) << 16),             // tensor_dim0[15:0] @ bits63:48
             (int)(((cols >> 16) & 0xFFFFu) |           // tensor_dim0[31:16]
                   ((rows & 0xFFFFu) << 16)),           // tensor_dim1[15:0]
             (int)(((rows >> 16) & 0xFFFFu) |           // tensor_dim1[31:16]
                   ((cols & 0xFFFFu) << 16)),           // tile_dim0
             (int)(rows & 0xFFFFu),                     // tile_dim1 (tile_dim2=0)
             (int)stride_elems,                         // tensor_dim0_stride[31:0]
             0, 0 };
  v4i z = {};
  TDM_CALL(g0, g1, z, z);
}

// ---------------- weights f32 -> bf16 ----------------
__global__ void k_cvt_w(const float* qw, const float* kw, const float* vw, const float* pw,
                        unsigned short* wq, unsigned short* wk,
                        unsigned short* wv, unsigned short* wp) {
  int i = blockIdx.x * blockDim.x + threadIdx.x;   // 512*512 threads
  wq[i] = f2bf(qw[i]); wk[i] = f2bf(kw[i]);
  wv[i] = f2bf(vw[i]); wp[i] = f2bf(pw[i]);
}

// ---------------- GroupNorm stats: one block per (b,g) ----------------
__global__ void k_gn_stats(const float* x, float* stats) {
  __shared__ float s1[256], s2[256];
  int bg = blockIdx.x;                               // b*32+g; group rows contiguous
  const float* p = x + (size_t)bg * (CPG * NPIX);
  float a = 0.f, q = 0.f;
  for (int j = threadIdx.x; j < CPG * NPIX; j += 256) { float v = p[j]; a += v; q += v * v; }
  s1[threadIdx.x] = a; s2[threadIdx.x] = q;
  __syncthreads();
  for (int off = 128; off > 0; off >>= 1) {
    if (threadIdx.x < off) { s1[threadIdx.x] += s1[threadIdx.x + off];
                             s2[threadIdx.x] += s2[threadIdx.x + off]; }
    __syncthreads();
  }
  if (threadIdx.x == 0) {
    const float inv = 1.f / (CPG * NPIX);
    float mean = s1[0] * inv;
    float var  = s2[0] * inv - mean * mean;
    stats[bg * 2]     = mean;
    stats[bg * 2 + 1] = rsqrtf(var + 1e-5f);
  }
}

// ---------------- GroupNorm apply + transpose -> h^T [B][HW][C] bf16 ----------------
__global__ void k_gn_apply(const float* x, const float* gw, const float* gb,
                           const float* stats, unsigned short* h_t) {
  __shared__ unsigned short t[64][72];               // [p_local][c_local], padded
  int p0 = blockIdx.x * 64, c0 = blockIdx.y * 64, b = blockIdx.z;
  int cl = threadIdx.x >> 2;                         // 0..63
  int pl = (threadIdx.x & 3) * 16;
  int c = c0 + cl, g = c >> 4;
  float mean = stats[(b * NGRP + g) * 2];
  float rstd = stats[(b * NGRP + g) * 2 + 1];
  float ga = gw[c], be = gb[c];
  const float* xr = x + ((size_t)b * CH + c) * NPIX + p0 + pl;
  for (int j = 0; j < 16; ++j)
    t[pl + j][cl] = f2bf((xr[j] - mean) * rstd * ga + be);
  __syncthreads();
  int pr = threadIdx.x >> 2;
  int chk = (threadIdx.x & 3) * 16;
  unsigned short* dst = h_t + ((size_t)b * NPIX + p0 + pr) * CH + c0 + chk;
  for (int j = 0; j < 16; ++j) dst[j] = t[pr][chk + j];
}

// ---------------- QKV GEMM: q^T,k^T [HW][C]; v [C][HW] ----------------
__global__ void __launch_bounds__(256)
k_qkv(const unsigned short* wq, const unsigned short* wk, const unsigned short* wv,
      const float* qb, const float* kb, const float* vb,
      const unsigned short* h_t,
      unsigned short* qT, unsigned short* kT, unsigned short* vv) {
  __shared__ unsigned short wt[16 * 512];
  __shared__ float bias[16];
  int pblk = blockIdx.x, ot = blockIdx.y;
  int b = blockIdx.z / 3, wsel = blockIdx.z % 3;
  const unsigned short* wsrc = wsel == 0 ? wq : (wsel == 1 ? wk : wv);
  const float* bsrc = wsel == 0 ? qb : (wsel == 1 ? kb : vb);
  if (threadIdx.x < 32) {                            // wave 0 issues the TDM load
    tdm_load_rows_bf16(wsrc + (size_t)ot * 16 * 512, wt, 16, 512, 512);
    __builtin_amdgcn_s_wait_tensorcnt(0);
  }
  if (threadIdx.x < 16) bias[threadIdx.x] = bsrc[ot * 16 + threadIdx.x];
  __syncthreads();
  int w = threadIdx.x >> 5, lane = threadIdx.x & 31;
  int hb = (lane >> 4) * 8;
  int p0 = pblk * 512 + w * 64;
  v8f zero = {};
  v8f acc[4] = {zero, zero, zero, zero};
  for (int cc = 0; cc < 16; ++cc) {
    v16bf a = frag_a(wt + cc * 32, 512, lane);
    for (int nt = 0; nt < 4; ++nt) {
      const unsigned short* bt = h_t + ((size_t)b * NPIX + p0 + nt * 16) * CH + cc * 32;
      acc[nt] = wmma_bf16(a, frag_b(bt, CH, lane), acc[nt]);
    }
  }
  for (int nt = 0; nt < 4; ++nt)
    for (int i = 0; i < 8; ++i) {
      int o = ot * 16 + i + hb;
      int p = p0 + nt * 16 + (lane & 15);
      float val = acc[nt][i] + bias[i + hb];
      if (wsel == 2)      vv[((size_t)b * CH + o) * NPIX + p] = f2bf(val);
      else if (wsel == 1) kT[((size_t)b * NPIX + p) * CH + o] = f2bf(val);
      else                qT[((size_t)b * NPIX + p) * CH + o] = f2bf(val);
    }
}

// ---------------- Flash attention: one block per 16-query tile ----------------
__global__ void __launch_bounds__(256)
k_attn(const unsigned short* qT, const unsigned short* kT,
       const unsigned short* vv, unsigned short* outT) {
  __shared__ unsigned short qs[16 * 512];   // query tile [16 q][512 c]
  __shared__ unsigned short ps[16 * 512];   // exp(S) tile [16 q][512 keys] bf16
  __shared__ float wred[8][16];
  __shared__ float mrow[16], lrow[16], frow[16];
  int qt = blockIdx.x, b = blockIdx.y;
  int w = threadIdx.x >> 5, lane = threadIdx.x & 31;
  int hb = (lane >> 4) * 8;
  if (threadIdx.x < 32) {                            // wave 0: TDM query-tile load
    tdm_load_rows_bf16(qT + ((size_t)b * NPIX + qt * 16) * CH, qs, 16, 512, 512);
    __builtin_amdgcn_s_wait_tensorcnt(0);
  }
  if (threadIdx.x < 16) { mrow[threadIdx.x] = -1e30f; lrow[threadIdx.x] = 0.f; }
  v8f zero = {};
  v8f oacc[4] = {zero, zero, zero, zero};   // wave's 16q x 64d output accumulator
  __syncthreads();
  const float scale = 0.044194173824159216f;          // 512^-0.5
  for (int kb2 = 0; kb2 < 8; ++kb2) {                 // 512-key blocks
    // ---- S = q^T k for this wave's 64 keys ----
    v8f sacc[4];
    for (int kt = 0; kt < 4; ++kt) {
      sacc[kt] = zero;
      int key0 = kb2 * 512 + w * 64 + kt * 16;
      const unsigned short* ktb = kT + ((size_t)b * NPIX + key0) * CH;
      __builtin_prefetch(ktb + 16 * CH, 0, 1);        // next key tile
      for (int cc = 0; cc < 16; ++cc) {
        v16bf a = frag_a(qs + cc * 32, 512, lane);
        sacc[kt] = wmma_bf16(a, frag_b(ktb + cc * 32, CH, lane), sacc[kt]);
      }
      for (int i = 0; i < 8; ++i) sacc[kt][i] *= scale;
    }
    // ---- running row-max ----
    float pm[8];
    for (int i = 0; i < 8; ++i)
      pm[i] = fmaxf(fmaxf(sacc[0][i], sacc[1][i]), fmaxf(sacc[2][i], sacc[3][i]));
    for (int off = 1; off < 16; off <<= 1)
      for (int i = 0; i < 8; ++i) pm[i] = fmaxf(pm[i], __shfl_xor(pm[i], off, 32));
    if ((lane & 15) == 0)
      for (int i = 0; i < 8; ++i) wred[w][i + hb] = pm[i];
    __syncthreads();
    if (threadIdx.x < 16) {
      float cm = wred[0][threadIdx.x];
      for (int j = 1; j < 8; ++j) cm = fmaxf(cm, wred[j][threadIdx.x]);
      float mo = mrow[threadIdx.x];
      float mn = fmaxf(mo, cm);
      mrow[threadIdx.x] = mn;
      frow[threadIdx.x] = __expf(mo - mn);
    }
    __syncthreads();
    // ---- P = exp(S - m), row-sum, stash bf16 P in LDS ----
    float mr[8], su[8];
    for (int i = 0; i < 8; ++i) { mr[i] = mrow[i + hb]; su[i] = 0.f; }
    for (int kt = 0; kt < 4; ++kt) {
      int keyl = w * 64 + kt * 16 + (lane & 15);
      for (int i = 0; i < 8; ++i) {
        float pv = __expf(sacc[kt][i] - mr[i]);
        su[i] += pv;
        ps[(i + hb) * 512 + keyl] = f2bf(pv);
      }
    }
    for (int off = 1; off < 16; off <<= 1)
      for (int i = 0; i < 8; ++i) su[i] += __shfl_xor(su[i], off, 32);
    if ((lane & 15) == 0)
      for (int i = 0; i < 8; ++i) wred[w][i + hb] = su[i];
    __syncthreads();
    if (threadIdx.x < 16) {
      float cs = 0.f;
      for (int j = 0; j < 8; ++j) cs += wred[j][threadIdx.x];
      lrow[threadIdx.x] = lrow[threadIdx.x] * frow[threadIdx.x] + cs;
    }
    __syncthreads();
    // ---- rescale O, then O += P @ V^T (wave owns 64 dims) ----
    float fr[8];
    for (int i = 0; i < 8; ++i) fr[i] = frow[i + hb];
    for (int dt = 0; dt < 4; ++dt)
      for (int i = 0; i < 8; ++i) oacc[dt][i] *= fr[i];
    for (int kc = 0; kc < 16; ++kc) {
      v16bf a = frag_a(ps + kc * 32, 512, lane);
      for (int dt = 0; dt < 4; ++dt) {
        int d0 = w * 64 + dt * 16;
        const unsigned short* bt = vv + ((size_t)b * CH + d0) * NPIX + kb2 * 512 + kc * 32;
        oacc[dt] = wmma_bf16(a, frag_b(bt, NPIX, lane), oacc[dt]);
      }
    }
    __syncthreads();
  }
  float il[8];
  for (int i = 0; i < 8; ++i) il[i] = 1.0f / lrow[i + hb];
  for (int dt = 0; dt < 4; ++dt)
    for (int i = 0; i < 8; ++i) {
      int q = qt * 16 + i + hb;
      int d = w * 64 + dt * 16 + (lane & 15);
      outT[((size_t)b * NPIX + q) * CH + d] = f2bf(oacc[dt][i] * il[i]);
    }
}

// ---------------- proj GEMM + bias + residual -> f32 out ----------------
__global__ void __launch_bounds__(256)
k_proj(const unsigned short* wp, const float* pb,
       const unsigned short* outT, const float* x, float* out) {
  __shared__ unsigned short wt[16 * 512];
  __shared__ float bias[16];
  int pblk = blockIdx.x, ot = blockIdx.y, b = blockIdx.z;
  if (threadIdx.x < 32) {                            // wave 0: TDM weight-tile load
    tdm_load_rows_bf16(wp + (size_t)ot * 16 * 512, wt, 16, 512, 512);
    __builtin_amdgcn_s_wait_tensorcnt(0);
  }
  if (threadIdx.x < 16) bias[threadIdx.x] = pb[ot * 16 + threadIdx.x];
  __syncthreads();
  int w = threadIdx.x >> 5, lane = threadIdx.x & 31;
  int hb = (lane >> 4) * 8;
  int p0 = pblk * 512 + w * 64;
  v8f zero = {};
  v8f acc[4] = {zero, zero, zero, zero};
  for (int cc = 0; cc < 16; ++cc) {
    v16bf a = frag_a(wt + cc * 32, 512, lane);
    for (int nt = 0; nt < 4; ++nt) {
      const unsigned short* bt = outT + ((size_t)b * NPIX + p0 + nt * 16) * CH + cc * 32;
      acc[nt] = wmma_bf16(a, frag_b(bt, CH, lane), acc[nt]);
    }
  }
  for (int nt = 0; nt < 4; ++nt)
    for (int i = 0; i < 8; ++i) {
      int o = ot * 16 + i + hb;
      int p = p0 + nt * 16 + (lane & 15);
      size_t idx = ((size_t)b * CH + o) * NPIX + p;
      out[idx] = x[idx] + bias[i + hb] + acc[nt][i];
    }
}

extern "C" void kernel_launch(void* const* d_in, const int* in_sizes, int n_in,
                              void* d_out, int out_size, void* d_ws, size_t ws_size,
                              hipStream_t stream) {
  (void)in_sizes; (void)n_in; (void)out_size; (void)ws_size;
  const float* x  = (const float*)d_in[0];
  const float* gw = (const float*)d_in[1];
  const float* gb = (const float*)d_in[2];
  const float* qw = (const float*)d_in[3];
  const float* qb = (const float*)d_in[4];
  const float* kw = (const float*)d_in[5];
  const float* kb = (const float*)d_in[6];
  const float* vw = (const float*)d_in[7];
  const float* vb = (const float*)d_in[8];
  const float* pw = (const float*)d_in[9];
  const float* pb = (const float*)d_in[10];
  float* out = (float*)d_out;
  char* ws = (char*)d_ws;

  const size_t WB = (size_t)CH * CH * 2;          // 512 KB per weight (bf16)
  unsigned short* wq = (unsigned short*)(ws);
  unsigned short* wk = (unsigned short*)(ws + WB);
  unsigned short* wv = (unsigned short*)(ws + 2 * WB);
  unsigned short* wp = (unsigned short*)(ws + 3 * WB);
  float* stats = (float*)(ws + 4 * WB);
  const size_t TB = (size_t)BATCH * NPIX * CH * 2; // 16 MB per bf16 tensor
  char* base = ws + 4 * WB + 4096;
  unsigned short* h_t = (unsigned short*)(base);
  unsigned short* qT  = (unsigned short*)(base + TB);
  unsigned short* kT  = (unsigned short*)(base + 2 * TB);
  unsigned short* vv  = (unsigned short*)(base + 3 * TB);
  unsigned short* oT  = (unsigned short*)(base + 4 * TB);

  k_cvt_w<<<(CH * CH) / 256, 256, 0, stream>>>(qw, kw, vw, pw, wq, wk, wv, wp);
  k_gn_stats<<<BATCH * NGRP, 256, 0, stream>>>(x, stats);
  k_gn_apply<<<dim3(NPIX / 64, CH / 64, BATCH), 256, 0, stream>>>(x, gw, gb, stats, h_t);
  k_qkv<<<dim3(NPIX / 512, CH / 16, 3 * BATCH), 256, 0, stream>>>(
      wq, wk, wv, qb, kb, vb, h_t, qT, kT, vv);
  k_attn<<<dim3(NPIX / 16, BATCH), 256, 0, stream>>>(qT, kT, vv, oT);
  k_proj<<<dim3(NPIX / 512, CH / 16, BATCH), 256, 0, stream>>>(wp, pb, oT, x, out);
}
